// Trans_WF_31748398252312
// MI455X (gfx1250) — matrix operand order repaired and checked by reading, not compiled, not run on hardware
//
#include <hip/hip_runtime.h>
#include <hip/hip_fp16.h>
#include <math.h>

typedef _Float16 h16;
typedef __attribute__((ext_vector_type(16))) _Float16 v16h;
typedef __attribute__((ext_vector_type(8)))  float    v8f;

#define EPSF 1e-5f

// ---------------------------------------------------------------------------
// roll(shift) + segment + f32->f16
// xr[(b*4+n)*2500+p] = x[b, (n*2500+p - shift) mod 10000]
// ---------------------------------------------------------------------------
__global__ void k_roll(const float* __restrict__ x, const int* __restrict__ shift,
                       h16* __restrict__ xr) {
  int i = blockIdx.x * 256 + threadIdx.x;
  if (i >= 128 * 10000) return;
  int b = i / 10000, l = i - b * 10000;
  int s = shift[0] % 10000; if (s < 0) s += 10000;
  int ls = l - s; if (ls < 0) ls += 10000;
  xr[i] = (h16)x[b * 10000 + ls];
}

// ---------------------------------------------------------------------------
// Block1 conv1 (Cin=1, k=7 'same') + BN + ReLU  (scalar path; K too small for WMMA)
// ---------------------------------------------------------------------------
__global__ void k_conv1_c1(const h16* __restrict__ xr, const float* __restrict__ W,
                           const float* __restrict__ Bb, const float* __restrict__ Gg,
                           const float* __restrict__ Be, const float* __restrict__ Mm,
                           const float* __restrict__ Vv, h16* __restrict__ dst) {
  int i = blockIdx.x * 256 + threadIdx.x;
  if (i >= 512 * 2500) return;
  int seg = i / 2500, p = i - seg * 2500;
  float in[7];
#pragma unroll
  for (int t = 0; t < 7; ++t) {
    int q = p + t - 3;
    in[t] = (q >= 0 && q < 2500) ? (float)xr[seg * 2500 + q] : 0.f;
  }
  for (int co = 0; co < 32; ++co) {
    float s = 0.f;
#pragma unroll
    for (int t = 0; t < 7; ++t) s += W[co * 7 + t] * in[t];
    float sc = Gg[co] * rsqrtf(Vv[co] + EPSF);
    float of = (Bb[co] - Mm[co]) * sc + Be[co];
    float u = fmaxf(s * sc + of, 0.f);
    dst[((size_t)seg * 32 + co) * 2500 + p] = (h16)u;
  }
}

// ---------------------------------------------------------------------------
// Implicit-GEMM conv1d k=7 'same' via V_WMMA_F32_16X16X32_F16.
//   out[co,p] = f( bn( sum_{t,ci} W[co,ci,t] * src[ci,p+t-3] ) , residual )
// RESMODE 0: out = relu(bn(conv))                    (first conv of a block)
// RESMODE 1: out = relu(relu(bn(conv)) + conv1x1(rsrc)+bd)   (WMMA residual)
// RESMODE 2: same but Cin_res==1 (scalar residual)
// Workgroup: 4 waves; each wave owns a 16(M=co) x 16(N=pos) C tile.
// LDS: Wl[16][Cs] per-tap weights (f16) + At[70][Cs] transposed act tile.
// ---------------------------------------------------------------------------
template<int RESMODE>
__global__ __launch_bounds__(128)
void k_conv7(const h16* __restrict__ src, int Cs, int L,
             const float* __restrict__ W,
             const float* __restrict__ Bb, const float* __restrict__ Gg,
             const float* __restrict__ Be, const float* __restrict__ Mm,
             const float* __restrict__ Vv,
             const h16* __restrict__ rsrc, int Cr,
             const float* __restrict__ Wd, const float* __restrict__ bd,
             h16* __restrict__ dst, int Cout) {
  __shared__ h16 lds[22016];              // 86 * 256 f16 = 44KB max
  const int tid  = threadIdx.x;
  const int wv   = tid >> 5;
  const int lane = tid & 31;
  const int hh   = lane >> 4;             // half-wave select
  const int nl   = lane & 15;
  const int p0   = blockIdx.x * 64;
  const int co0  = blockIdx.y * 16;
  const int seg  = blockIdx.z;

  h16* Wl = lds;                          // [16][Cs]
  h16* At = lds + 16 * Cs;                // [70][Cs] positions p0-3 .. p0+66
  const h16* sseg = src + (size_t)seg * Cs * L;

  for (int idx = tid; idx < 70 * Cs; idx += 128) {
    int j = idx / Cs, ci = idx - j * Cs;
    int p = p0 - 3 + j;
    h16 v = (h16)0.f;
    if (p >= 0 && p < L) v = sseg[(size_t)ci * L + p];
    At[j * Cs + ci] = v;
  }

  v8f acc = {};
  for (int t = 0; t < 7; ++t) {
    __syncthreads();
    for (int idx = tid; idx < 16 * Cs; idx += 128) {   // stage tap-t weights as f16
      int m = idx / Cs, ci = idx - m * Cs;
      Wl[m * Cs + ci] = (h16)W[((size_t)(co0 + m) * Cs + ci) * 7 + t];
    }
    __syncthreads();
    const int j = wv * 16 + nl + t;                    // input position row in At
    for (int ci0 = 0; ci0 < Cs; ci0 += 32) {
      v16h a, b;
      // A 16x32 f16 layout: lane m=nl, half hh; e -> k=hh*16+e
      const h16* ar = Wl + nl * Cs + ci0 + hh * 16;
#pragma unroll
      for (int e = 0; e < 8; ++e) { a[e] = ar[e]; a[8 + e] = ar[8 + e]; }
      // B 32x16 f16 layout: lane col n=nl, half hh; e -> k=hh*16+e
      const h16* br = At + j * Cs + ci0 + hh * 16;
#pragma unroll
      for (int e = 0; e < 16; ++e) b[e] = br[e];
      acc = __builtin_amdgcn_wmma_f32_16x16x32_f16(false, a, false, b,
                                                   (short)0, acc, false, false);
    }
  }

  v8f accr = {};
  if constexpr (RESMODE == 1) {           // 1x1 residual conv as second WMMA GEMM
    __syncthreads();
    h16* Wdl = lds;                       // [16][Cr]
    h16* Rt  = lds + 16 * Cr;             // [64][Cr]
    const h16* rseg = rsrc + (size_t)seg * Cr * L;
    for (int idx = tid; idx < 16 * Cr; idx += 128) {
      int m = idx / Cr, ci = idx - m * Cr;
      Wdl[m * Cr + ci] = (h16)Wd[(size_t)(co0 + m) * Cr + ci];
    }
    for (int idx = tid; idx < 64 * Cr; idx += 128) {
      int j = idx / Cr, ci = idx - j * Cr;
      int p = p0 + j;
      Rt[j * Cr + ci] = (p < L) ? rseg[(size_t)ci * L + p] : (h16)0.f;
    }
    __syncthreads();
    const int j = wv * 16 + nl;
    for (int ci0 = 0; ci0 < Cr; ci0 += 32) {
      v16h a, b;
      const h16* ar = Wdl + nl * Cr + ci0 + hh * 16;
#pragma unroll
      for (int e = 0; e < 8; ++e) { a[e] = ar[e]; a[8 + e] = ar[8 + e]; }
      const h16* br = Rt + j * Cr + ci0 + hh * 16;
#pragma unroll
      for (int e = 0; e < 16; ++e) b[e] = br[e];
      accr = __builtin_amdgcn_wmma_f32_16x16x32_f16(false, a, false, b,
                                                    (short)0, accr, false, false);
    }
  }

  // C/D layout: lane l, vgpr r -> row M = r + 8*(l/16), col N = l%16
  const int p = p0 + wv * 16 + nl;
  if (p < L) {
#pragma unroll
    for (int r = 0; r < 8; ++r) {
      int co = co0 + r + 8 * hh;
      float sc = Gg[co] * rsqrtf(Vv[co] + EPSF);
      float of = (Bb[co] - Mm[co]) * sc + Be[co];
      float u = fmaxf(acc[r] * sc + of, 0.f);
      if constexpr (RESMODE == 1) {
        u = fmaxf(u + accr[r] + bd[co], 0.f);
      } else if constexpr (RESMODE == 2) {
        float xs = (float)rsrc[(size_t)seg * L + p];
        u = fmaxf(u + Wd[co] * xs + bd[co], 0.f);
      }
      dst[((size_t)seg * Cout + co) * L + p] = (h16)u;
    }
  }
}

// ---------------------------------------------------------------------------
// MaxPool1d(kernel=8, stride=4), VALID
// ---------------------------------------------------------------------------
__global__ void k_pool(const h16* __restrict__ src, h16* __restrict__ dst,
                       int Lin, int Lout, int total) {
  int i = blockIdx.x * 256 + threadIdx.x;
  if (i >= total) return;
  int q = i % Lout, sc = i / Lout;        // sc = seg*C + c
  const h16* s = src + (size_t)sc * Lin + 4 * q;
  float m = (float)s[0];
#pragma unroll
  for (int j = 1; j < 8; ++j) m = fmaxf(m, (float)s[j]);
  dst[i] = (h16)m;
}

// ---------------------------------------------------------------------------
// Tokens: [512,256,8] -> [128,32,256] + pos
// ---------------------------------------------------------------------------
__global__ void k_tok(const h16* __restrict__ a4, const float* __restrict__ pos,
                      float* __restrict__ tok) {
  int i = blockIdx.x * 256 + threadIdx.x;
  if (i >= 128 * 32 * 256) return;
  int d = i & 255, t = (i >> 8) & 31, b = i >> 13;
  int ns = t >> 3, j = t & 7;
  tok[i] = (float)a4[(((size_t)(b * 4 + ns)) * 256 + d) * 8 + j] + pos[t * 256 + d];
}

// ---------------------------------------------------------------------------
// LayerNorm over D=256, one token per 256-thread block; f16 out
// ---------------------------------------------------------------------------
__global__ __launch_bounds__(256)
void k_ln(const float* __restrict__ x, const float* __restrict__ g,
          const float* __restrict__ b, h16* __restrict__ y) {
  __shared__ float red[256];
  int tok = blockIdx.x, d = threadIdx.x;
  float v = x[(size_t)tok * 256 + d];
  red[d] = v; __syncthreads();
  for (int s2 = 128; s2 > 0; s2 >>= 1) { if (d < s2) red[d] += red[d + s2]; __syncthreads(); }
  float mu = red[0] * (1.f / 256.f);
  __syncthreads();
  float dv = v - mu;
  red[d] = dv * dv; __syncthreads();
  for (int s2 = 128; s2 > 0; s2 >>= 1) { if (d < s2) red[d] += red[d + s2]; __syncthreads(); }
  float rstd = rsqrtf(red[0] * (1.f / 256.f) + EPSF);
  y[(size_t)tok * 256 + d] = (h16)(dv * rstd * g[d] + b[d]);
}

// ---------------------------------------------------------------------------
// WMMA GEMM: C[M,N] = A[M,K](f16) x Bw[K,N](f32->f16) ; 4 waves = 64 N per block
// MODE 0: outf = acc + bias (f32)
// MODE 1: resid += acc + bias (residual add, in place, f32)
// MODE 2: outh = gelu_exact(acc + bias) (f16)
// ---------------------------------------------------------------------------
template<int MODE>
__global__ __launch_bounds__(128)
void k_gemm(const h16* __restrict__ A, const float* __restrict__ Bw,
            const float* __restrict__ bias, float* __restrict__ resid,
            float* __restrict__ outf, h16* __restrict__ outh, int N, int K) {
  __shared__ h16 At[16 * 64];
  __shared__ h16 Bt[64 * 64];
  const int tid = threadIdx.x, wv = tid >> 5, lane = tid & 31;
  const int hh = lane >> 4, nl = lane & 15;
  const int m0 = blockIdx.y * 16;
  const int nb = blockIdx.x * 64;
  v8f acc = {};
  for (int k0 = 0; k0 < K; k0 += 64) {
    __syncthreads();
    for (int idx = tid; idx < 16 * 64; idx += 128) {
      int m = idx >> 6, kk = idx & 63;
      At[idx] = A[(size_t)(m0 + m) * K + k0 + kk];
    }
    for (int idx = tid; idx < 64 * 64; idx += 128) {
      int n = idx >> 6, kk = idx & 63;
      Bt[idx] = (h16)Bw[(size_t)(k0 + kk) * N + nb + n];   // transpose to [n][k]
    }
    __syncthreads();
#pragma unroll
    for (int kc = 0; kc < 64; kc += 32) {
      v16h a, b;
      const h16* ar = At + nl * 64 + kc + hh * 16;
#pragma unroll
      for (int e = 0; e < 8; ++e) { a[e] = ar[e]; a[8 + e] = ar[8 + e]; }
      const h16* br = Bt + (wv * 16 + nl) * 64 + kc + hh * 16;
#pragma unroll
      for (int e = 0; e < 16; ++e) b[e] = br[e];
      acc = __builtin_amdgcn_wmma_f32_16x16x32_f16(false, a, false, b,
                                                   (short)0, acc, false, false);
    }
  }
  const int n = nb + wv * 16 + nl;
#pragma unroll
  for (int r = 0; r < 8; ++r) {
    int m = m0 + r + 8 * hh;
    float v = acc[r] + bias[n];
    if constexpr (MODE == 0) {
      outf[(size_t)m * N + n] = v;
    } else if constexpr (MODE == 1) {
      size_t o = (size_t)m * N + n;
      resid[o] = resid[o] + v;
    } else {
      outh[(size_t)m * N + n] = (h16)(0.5f * v * (1.f + erff(v * 0.70710678118654752f)));
    }
  }
}

// ---------------------------------------------------------------------------
// Attention (N=32, HD=32, H=8): tiny FLOPs, VALU path.
// ---------------------------------------------------------------------------
__global__ void k_attn_scores(const float* __restrict__ qkv, float* __restrict__ att) {
  int bh = blockIdx.x, b = bh >> 3, hh = bh & 7;
  const float* base = qkv + (size_t)b * 32 * 768;
  for (int e = threadIdx.x; e < 1024; e += 256) {
    int n = e >> 5, m = e & 31;
    const float* q = base + n * 768 + hh * 32;
    const float* k = base + m * 768 + 256 + hh * 32;
    float s = 0.f;
#pragma unroll
    for (int d = 0; d < 32; ++d) s += q[d] * k[d];
    att[(size_t)bh * 1024 + e] = s * 0.17677669529663689f;   // 32^-0.5
  }
}

__global__ void k_attn_soft(float* __restrict__ att) {      // top-20 mask + softmax
  int row = blockIdx.x * 256 + threadIdx.x;
  if (row >= 32768) return;
  float* a = att + (size_t)row * 32;
  float v[32], c[32];
  for (int i = 0; i < 32; ++i) { v[i] = a[i]; c[i] = v[i]; }
  float thr = -1e30f;
  for (int it = 0; it < 20; ++it) {                          // 20th largest
    int bi = 0; float bv = -1e30f;
    for (int i = 0; i < 32; ++i) if (c[i] > bv) { bv = c[i]; bi = i; }
    thr = bv; c[bi] = -1e30f;
  }
  float mx = -1e30f;
  for (int i = 0; i < 32; ++i) if (v[i] >= thr && v[i] > mx) mx = v[i];
  float sum = 0.f;
  for (int i = 0; i < 32; ++i) {
    float e = (v[i] >= thr) ? expf(v[i] - mx) : 0.f;
    v[i] = e; sum += e;
  }
  float inv = 1.f / sum;
  for (int i = 0; i < 32; ++i) a[i] = v[i] * inv;
}

__global__ void k_attn_out(const float* __restrict__ qkv, const float* __restrict__ att,
                           h16* __restrict__ o16) {
  int bh = blockIdx.x, b = bh >> 3, hh = bh & 7;
  for (int e = threadIdx.x; e < 1024; e += 256) {
    int n = e >> 5, hd = e & 31;
    const float* ar = att + (size_t)bh * 1024 + n * 32;
    const float* vb = qkv + (size_t)b * 32 * 768 + 512 + hh * 32 + hd;
    float s = 0.f;
#pragma unroll
    for (int m = 0; m < 32; ++m) s += ar[m] * vb[m * 768];
    o16[((size_t)b * 32 + n) * 256 + hh * 32 + hd] = (h16)s;
  }
}

__global__ void k_mean(const float* __restrict__ tok, float* __restrict__ out) {
  int i = blockIdx.x * 256 + threadIdx.x;
  if (i >= 128 * 256) return;
  int b = i >> 8, d = i & 255;
  float s = 0.f;
  for (int t = 0; t < 32; ++t) s += tok[((size_t)b * 32 + t) * 256 + d];
  out[i] = s * (1.f / 32.f);
}

// ---------------------------------------------------------------------------
// Host orchestration
// ---------------------------------------------------------------------------
extern "C" void kernel_launch(void* const* d_in, const int* in_sizes, int n_in,
                              void* d_out, int out_size, void* d_ws, size_t ws_size,
                              hipStream_t stream) {
  (void)out_size; (void)ws_size;
  // locate x (1,280,000 elems) and shift (1 elem) by size; rest are param leaves
  int xi = -1, si = -1;
  for (int i = 0; i < n_in; ++i) {
    if (in_sizes[i] == 1280000) xi = i;
    else if (in_sizes[i] == 1) si = i;
  }
  const float* X = (const float*)d_in[xi];
  const int* SHIFT = (const int*)d_in[si];
  const float* P[96]; int Psz[96]; int np = 0;
  for (int i = 0; i < n_in; ++i)
    if (i != xi && i != si) { P[np] = (const float*)d_in[i]; Psz[np] = in_sizes[i]; ++np; }
  // first block leaf: 224 (=w1, insertion order) vs 32 (=b1, jax alphabetical)
  const bool ins = (np > 0 && Psz[0] > 200);
  // canonical [w1,b1,g1,be1,m1,v1,w2,b2,g2,be2,m2,v2,wd,bd] -> alphabetical slot
  static const int MB_A[14] = {11, 0, 5, 3, 7, 9, 12, 1, 6, 4, 8, 10, 13, 2};
  // canonical [ln1_g,ln1_b,qkv_w,qkv_b,proj_w,proj_b,ln2_g,ln2_b,fc1_w,fc1_b,fc2_w,fc2_b]
  static const int ML_A[12] = {5, 4, 11, 10, 9, 8, 7, 6, 1, 0, 3, 2};
  enum { W1, B1, G1, BE1, M1, V1, W2, B2, G2, BE2, M2, V2, WD, BD };
  enum { LN1G, LN1B, QKVW, QKVB, PROJW, PROJB, LN2G, LN2B, FC1W, FC1B, FC2W, FC2B };
  auto BP = [&](int blk, int w) -> const float* { return P[blk * 14 + (ins ? w : MB_A[w])]; };
  auto LP = [&](int l, int w) -> const float*  { return P[56 + l * 12 + (ins ? w : ML_A[w])]; };
  const float* POS = P[80];

  char* ws = (char*)d_ws;
  h16*   XR   = (h16*)(ws);                           // 512*2500 f16
  h16*   MID  = (h16*)(ws + ((size_t)6   << 20));     // <= 512*32*2500 f16
  h16*   FULL = (h16*)(ws + ((size_t)90  << 20));     // <= 512*32*2500 f16
  h16*   ACTA = (h16*)(ws + ((size_t)174 << 20));     // act1 / act3
  h16*   ACTB = (h16*)(ws + ((size_t)196 << 20));     // act2 / act4
  float* TOK  = (float*)(ws + ((size_t)208 << 20));   // 4096*256 f32
  h16*   YH   = (h16*)(ws + ((size_t)213 << 20));     // 4096*256 f16
  float* QKV  = (float*)(ws + ((size_t)216 << 20));   // 4096*768 f32
  float* ATT  = (float*)(ws + ((size_t)229 << 20));   // 128*8*32*32 f32
  h16*   O16  = (h16*)(ws + ((size_t)234 << 20));     // 4096*256 f16
  h16*   H16  = (h16*)(ws + ((size_t)237 << 20));     // 4096*1024 f16

  // ---- conv stem ----
  k_roll<<<5000, 256, 0, stream>>>(X, SHIFT, XR);
  k_conv1_c1<<<5000, 256, 0, stream>>>(XR, BP(0, W1), BP(0, B1), BP(0, G1), BP(0, BE1),
                                       BP(0, M1), BP(0, V1), MID);
  k_conv7<2><<<dim3(40, 2, 512), 128, 0, stream>>>(MID, 32, 2500, BP(0, W2),
      BP(0, B2), BP(0, G2), BP(0, BE2), BP(0, M2), BP(0, V2),
      XR, 1, BP(0, WD), BP(0, BD), FULL, 32);
  k_pool<<<(512 * 32 * 624 + 255) / 256, 256, 0, stream>>>(FULL, ACTA, 2500, 624, 512 * 32 * 624);

  k_conv7<0><<<dim3(10, 4, 512), 128, 0, stream>>>(ACTA, 32, 624, BP(1, W1),
      BP(1, B1), BP(1, G1), BP(1, BE1), BP(1, M1), BP(1, V1),
      nullptr, 0, nullptr, nullptr, MID, 64);
  k_conv7<1><<<dim3(10, 4, 512), 128, 0, stream>>>(MID, 64, 624, BP(1, W2),
      BP(1, B2), BP(1, G2), BP(1, BE2), BP(1, M2), BP(1, V2),
      ACTA, 32, BP(1, WD), BP(1, BD), FULL, 64);
  k_pool<<<(512 * 64 * 155 + 255) / 256, 256, 0, stream>>>(FULL, ACTB, 624, 155, 512 * 64 * 155);

  k_conv7<0><<<dim3(3, 8, 512), 128, 0, stream>>>(ACTB, 64, 155, BP(2, W1),
      BP(2, B1), BP(2, G1), BP(2, BE1), BP(2, M1), BP(2, V1),
      nullptr, 0, nullptr, nullptr, MID, 128);
  k_conv7<1><<<dim3(3, 8, 512), 128, 0, stream>>>(MID, 128, 155, BP(2, W2),
      BP(2, B2), BP(2, G2), BP(2, BE2), BP(2, M2), BP(2, V2),
      ACTB, 64, BP(2, WD), BP(2, BD), FULL, 128);
  k_pool<<<(512 * 128 * 37 + 255) / 256, 256, 0, stream>>>(FULL, ACTA, 155, 37, 512 * 128 * 37);

  k_conv7<0><<<dim3(1, 16, 512), 128, 0, stream>>>(ACTA, 128, 37, BP(3, W1),
      BP(3, B1), BP(3, G1), BP(3, BE1), BP(3, M1), BP(3, V1),
      nullptr, 0, nullptr, nullptr, MID, 256);
  k_conv7<1><<<dim3(1, 16, 512), 128, 0, stream>>>(MID, 256, 37, BP(3, W2),
      BP(3, B2), BP(3, G2), BP(3, BE2), BP(3, M2), BP(3, V2),
      ACTA, 128, BP(3, WD), BP(3, BD), FULL, 256);
  k_pool<<<(512 * 256 * 8 + 255) / 256, 256, 0, stream>>>(FULL, ACTB, 37, 8, 512 * 256 * 8);

  // ---- transformer ----
  k_tok<<<4096, 256, 0, stream>>>(ACTB, POS, TOK);
  for (int l = 0; l < 2; ++l) {
    k_ln<<<4096, 256, 0, stream>>>(TOK, LP(l, LN1G), LP(l, LN1B), YH);
    k_gemm<0><<<dim3(12, 256), 128, 0, stream>>>(YH, LP(l, QKVW), LP(l, QKVB),
                                                 nullptr, QKV, nullptr, 768, 256);
    k_attn_scores<<<1024, 256, 0, stream>>>(QKV, ATT);
    k_attn_soft<<<128, 256, 0, stream>>>(ATT);
    k_attn_out<<<1024, 256, 0, stream>>>(QKV, ATT, O16);
    k_gemm<1><<<dim3(4, 256), 128, 0, stream>>>(O16, LP(l, PROJW), LP(l, PROJB),
                                                TOK, nullptr, nullptr, 256, 256);
    k_ln<<<4096, 256, 0, stream>>>(TOK, LP(l, LN2G), LP(l, LN2B), YH);
    k_gemm<2><<<dim3(16, 256), 128, 0, stream>>>(YH, LP(l, FC1W), LP(l, FC1B),
                                                 nullptr, nullptr, H16, 1024, 256);
    k_gemm<1><<<dim3(4, 256), 128, 0, stream>>>(H16, LP(l, FC2W), LP(l, FC2B),
                                                TOK, nullptr, nullptr, 256, 1024);
  }
  k_mean<<<128, 256, 0, stream>>>(TOK, (float*)d_out);
}
